// decoderBlock_sparse_22110491640060
// MI455X (gfx1250) — matrix operand order repaired and checked by reading, not compiled, not run on hardware
//
#include <hip/hip_runtime.h>

// ============================================================
// MI455X (gfx1250, wave32) implementation.
// Activations: channel-last bf16 [voxel][64]. Weights: pre-packed
// bf16 B-fragments. GEMMs via v_wmma_f32_16x16x32_bf16.
// CDNA5 data movement: TDM tensor_load_to_lds for weight staging,
// global_load_async_to_lds_b128 for activation staging (ASYNCcnt),
// s_wait_tensorcnt / s_wait_asynccnt for completion.
// ============================================================

typedef __bf16 bf16;
typedef __attribute__((ext_vector_type(16))) __bf16 v16bf;
typedef __attribute__((ext_vector_type(8)))  __bf16 bf16x8;
typedef __attribute__((ext_vector_type(8)))  float  v8f;
typedef __attribute__((ext_vector_type(4)))  unsigned su4;
typedef __attribute__((ext_vector_type(8)))  unsigned su8;

#define DD 48
#define HH 64
#define WW 128
#define NV (DD*HH*WW)          // 393216 voxels
#define NCH 64
#define NBLK3 1536             // conv3: 12288 waves x 32 voxels
#define NBLK1 3072             // conv1: 24576 waves x 16 voxels

union AB32 { v16bf v; bf16x8 h[2]; unsigned long long q[4]; };

__device__ __forceinline__ bf16 f2bf(float x){
  unsigned u = __builtin_bit_cast(unsigned, x);
  unsigned r = u + 0x7FFFu + ((u >> 16) & 1u);   // round-to-nearest-even
  unsigned short s = (unsigned short)(r >> 16);
  return __builtin_bit_cast(bf16, s);
}
__device__ __forceinline__ float bf2f(bf16 x){
  unsigned short s = __builtin_bit_cast(unsigned short, x);
  unsigned u = ((unsigned)s) << 16;
  return __builtin_bit_cast(float, u);
}
__device__ __forceinline__ v16bf zeroA(){
  AB32 u; u.q[0]=0ull; u.q[1]=0ull; u.q[2]=0ull; u.q[3]=0ull; return u.v;
}
// A fragment (16x32 bf16): lane half h holds e=0..15 -> K=h*16+e.
// rowbase points at K=0 of this lane's row.
__device__ __forceinline__ v16bf loadA(const bf16* rowbase, int half){
  AB32 u;
  u.h[0] = *reinterpret_cast<const bf16x8*>(rowbase + half*16);
  u.h[1] = *reinterpret_cast<const bf16x8*>(rowbase + half*16 + 8);
  return u.v;
}
// B fragment: pre-packed; each lane reads 16 contiguous bf16.
__device__ __forceinline__ v16bf loadB(const bf16* p){
  AB32 u;
  u.h[0] = *reinterpret_cast<const bf16x8*>(p);
  u.h[1] = *reinterpret_cast<const bf16x8*>(p + 8);
  return u.v;
}
__device__ __forceinline__ v8f zero8(){
  v8f z;
#pragma unroll
  for (int i = 0; i < 8; ++i) z[i] = 0.f;
  return z;
}

#define WMMA_BF16(A,B,C) \
  __builtin_amdgcn_wmma_f32_16x16x32_bf16(false,(A),false,(B),(short)0,(C),false,false)

// --- TDM: 1-D tile of `nd` dwords, global -> LDS (D# per ISA 08 sec 8) ---
__device__ __forceinline__ void tdm_load_1d(unsigned lds_off, const void* gptr,
                                            unsigned nd){
  unsigned long long ga = (unsigned long long)(size_t)gptr;
  su4 g0;
  g0[0] = 1u;                                        // count=1 (user D#)
  g0[1] = lds_off;                                   // lds_addr (bytes)
  g0[2] = (unsigned)ga;                              // global_addr[31:0]
  g0[3] = ((unsigned)(ga >> 32) & 0x1FFFFFFu) | (2u << 30); // [56:32] | type=2
  su8 g1;
  g1[0] = 2u << 16;                  // workgroup_mask=0, data_size=4B
  g1[1] = (nd & 0xFFFFu) << 16;      // tensor_dim0[15:0] in bits[63:48]
  g1[2] = ((nd >> 16) & 0xFFFFu) | (1u << 16); // dim0 hi | tensor_dim1=1
  g1[3] = (nd & 0xFFFFu) << 16;      // tile_dim0 = nd (bits 127:112)
  g1[4] = 1u;                        // tile_dim1=1, tile_dim2=0
  g1[5] = nd;                        // tensor_dim0_stride
  g1[6] = 0u;
  g1[7] = 0u;
  asm volatile("tensor_load_to_lds %0, %1" :: "s"(g0), "s"(g1) : "memory");
}

// --- async per-lane 16B copy, global -> LDS (ASYNCcnt) ---
__device__ __forceinline__ void async_b128(unsigned lds_off, const void* gptr){
  unsigned long long ga = (unsigned long long)(size_t)gptr;
  asm volatile("global_load_async_to_lds_b128 %0, %1, off"
               :: "v"(lds_off), "v"(ga) : "memory");
}
__device__ __forceinline__ void wait_async0(){
  asm volatile("s_wait_asynccnt 0x0" ::: "memory");
}

// ---------------- layout conversion: NCDHW f32 -> [vox][ch] bf16 ----
__global__ void cvt_in(const float* __restrict__ src, bf16* __restrict__ dst){
  long i = (long)blockIdx.x*blockDim.x + threadIdx.x;
  if (i >= (long)NV*NCH) return;
  long c = i / NV;
  long v = i - c*NV;
  dst[v*NCH + c] = f2bf(src[i]);
}

// ---------------- weight pack: OIDHW f32 -> B-fragment bf16 ---------
// dst[((nb*ksteps + kstep)*32 + lane)*16 + e], K = kstep*32+(lane>>4)*16+e,
// O = nb*16 + (lane&15). K-ordering: K = tap*64+cin (is3) else K = cin.
__global__ void pack_w(const float* __restrict__ w, bf16* __restrict__ dst,
                       int ksteps, int is3){
  int i = blockIdx.x*blockDim.x + threadIdx.x;
  int total = 4*ksteps*32*16;
  if (i >= total) return;
  int e     = i & 15;
  int lane  = (i >> 4) & 31;
  int rest  = i >> 9;
  int kstep = rest % ksteps;
  int nb    = rest / ksteps;
  int K = kstep*32 + (lane >> 4)*16 + e;
  int o = nb*16 + (lane & 15);
  float v;
  if (is3) { int t = K / 64, c = K % 64; v = w[(o*64 + c)*27 + t]; }
  else     { v = w[o*64 + K]; }
  dst[i] = f2bf(v);
}

// ---------------- 3x3x3 conv, implicit GEMM, WMMA, M=32/wave --------
// MODE 0: plain; MODE 1: + fused 1x1 side path; MODE 2: + ReLU store.
template <int MODE>
__global__ __launch_bounds__(256) void conv3_wmma(
    const bf16* __restrict__ act,      // [NV][64]
    const bf16* __restrict__ wp,       // [4][54][32][16] packed
    const bf16* __restrict__ act_side, // MODE==1
    const bf16* __restrict__ wp_side,  // MODE==1: [4][2][32][16]
    bf16* __restrict__ out)            // [NV][64]
{
  const int lane = threadIdx.x & 31;
  const int wave = blockIdx.x*8 + (threadIdx.x >> 5);  // 12288 waves
  const int half = lane >> 4;
  const int m    = lane & 15;
  const int tw = wave & 3, th = (wave >> 2) & 63, td = wave >> 8;
  const int w0 = tw*32;                 // 32-voxel strip along W
  const int vox0 = (td*HH + th)*WW + w0;

  __builtin_prefetch(act + (long)vox0*NCH, 0, 3);   // global_prefetch_b8

  v8f acc[8];                            // [strip(2)][nblk(4)]
#pragma unroll
  for (int i = 0; i < 8; ++i) acc[i] = zero8();

  int t = 0;
  for (int dz = -1; dz <= 1; ++dz){
    const int zz = td + dz;
    const bool zok = (unsigned)zz < (unsigned)DD;
    for (int dy = -1; dy <= 1; ++dy){
      const int yy = th + dy;
      const bool plok = zok && ((unsigned)yy < (unsigned)HH);
      const int rowv = (plok ? (zz*HH + yy) : 0) * WW;
      for (int dx = -1; dx <= 1; ++dx, ++t){
        const int x0 = w0 + m + dx;
        const int x1 = x0 + 16;
        const bool ok0 = plok && ((unsigned)x0 < (unsigned)WW);
        const bool ok1 = plok && ((unsigned)x1 < (unsigned)WW);
        const bf16* rb0 = act + (long)(rowv + (ok0 ? x0 : 0))*NCH;
        const bf16* rb1 = act + (long)(rowv + (ok1 ? x1 : 0))*NCH;
#pragma unroll
        for (int kh = 0; kh < 2; ++kh){
          v16bf a0 = ok0 ? loadA(rb0 + kh*32, half) : zeroA();
          v16bf a1 = ok1 ? loadA(rb1 + kh*32, half) : zeroA();
#pragma unroll
          for (int nb = 0; nb < 4; ++nb){
            v16bf b = loadB(wp + ((nb*54 + (t*2 + kh))*32 + lane)*16);
            acc[nb]     = WMMA_BF16(a0, b, acc[nb]);
            acc[4 + nb] = WMMA_BF16(a1, b, acc[4 + nb]);
          }
        }
      }
    }
  }
  if (MODE == 1){
    const bf16* rb0 = act_side + (long)(vox0 + m)*NCH;
    const bf16* rb1 = act_side + (long)(vox0 + 16 + m)*NCH;
#pragma unroll
    for (int kh = 0; kh < 2; ++kh){
      v16bf a0 = loadA(rb0 + kh*32, half);
      v16bf a1 = loadA(rb1 + kh*32, half);
#pragma unroll
      for (int nb = 0; nb < 4; ++nb){
        v16bf b = loadB(wp_side + ((nb*2 + kh)*32 + lane)*16);
        acc[nb]     = WMMA_BF16(a0, b, acc[nb]);
        acc[4 + nb] = WMMA_BF16(a1, b, acc[4 + nb]);
      }
    }
  }
  // C layout: VGPR r -> M = half*8 + r ; N = lane&15 (+16*nb)
  const int n0 = lane & 15;
#pragma unroll
  for (int s = 0; s < 2; ++s)
#pragma unroll
    for (int nb = 0; nb < 4; ++nb)
#pragma unroll
      for (int r = 0; r < 8; ++r){
        float v = acc[s*4 + nb][r];
        if (MODE == 2) v = fmaxf(v, 0.f);
        out[(long)(vox0 + s*16 + half*8 + r)*NCH + nb*16 + n0] = f2bf(v);
      }
}

// ------- fused 1x1 residual block: relu(x1 + W2*x1), x1 = W1*x -------
// Weights staged per-block by the TDM; activations staged per-wave by
// async global->LDS copies (contiguous 2KB per wave).
__global__ __launch_bounds__(256) void conv1_res_wmma(
    const bf16* __restrict__ act, const bf16* __restrict__ w1p,
    const bf16* __restrict__ w2p, bf16* __restrict__ out)
{
  __shared__ __align__(16) bf16 wstage[2][4096];     // 16 KB: W1,W2 packs (TDM)
  __shared__ __align__(16) bf16 astage[8][16*NCH];   // 16 KB: input acts (async)
  __shared__ __align__(16) bf16 xstage[8][16*NCH];   // 16 KB: x1 bounce
  const int lane = threadIdx.x & 31;
  const int wv   = threadIdx.x >> 5;
  const int wave = blockIdx.x*8 + wv;
  const int half = lane >> 4, m = lane & 15, n0 = lane & 15;
  const int vox0 = wave*16;

  // --- stage this wave's 2KB activation block: 32 lanes x 4 x 16B ---
  {
    const char* src = (const char*)(act + (long)vox0*NCH);
    unsigned lbase = (unsigned)(size_t)&astage[wv][0];
#pragma unroll
    for (int i = 0; i < 4; ++i){
      unsigned off = (unsigned)lane*16u + (unsigned)i*512u;
      async_b128(lbase + off, src + off);
    }
  }
  // --- wave 0: TDM-stage both weight packs (2048 dwords each) ---
  if (wv == 0){
    tdm_load_1d((unsigned)(size_t)&wstage[0][0], w1p, 2048u);
    tdm_load_1d((unsigned)(size_t)&wstage[1][0], w2p, 2048u);
  }
  wait_async0();                          // this wave's ASYNCcnt -> 0
  if (wv == 0) __builtin_amdgcn_s_wait_tensorcnt(0);
  __syncthreads();                        // weights visible to all waves

  v8f acc1[4];
#pragma unroll
  for (int nb = 0; nb < 4; ++nb) acc1[nb] = zero8();
#pragma unroll
  for (int kh = 0; kh < 2; ++kh){
    v16bf a = loadA(&astage[wv][m*NCH + kh*32], half);
#pragma unroll
    for (int nb = 0; nb < 4; ++nb){
      v16bf b = loadB(&wstage[0][((nb*2 + kh)*32 + lane)*16]);
      acc1[nb] = WMMA_BF16(a, b, acc1[nb]);
    }
  }
  // x1 -> LDS (transposes C-layout into A-layout rows)
#pragma unroll
  for (int nb = 0; nb < 4; ++nb)
#pragma unroll
    for (int r = 0; r < 8; ++r)
      xstage[wv][(half*8 + r)*NCH + nb*16 + n0] = f2bf(acc1[nb][r]);
  __syncthreads();

  v8f acc2[4];
#pragma unroll
  for (int nb = 0; nb < 4; ++nb) acc2[nb] = zero8();
#pragma unroll
  for (int kh = 0; kh < 2; ++kh){
    v16bf a = loadA(&xstage[wv][m*NCH + kh*32], half);
#pragma unroll
    for (int nb = 0; nb < 4; ++nb){
      v16bf b = loadB(&wstage[1][((nb*2 + kh)*32 + lane)*16]);
      acc2[nb] = WMMA_BF16(a, b, acc2[nb]);
    }
  }
#pragma unroll
  for (int nb = 0; nb < 4; ++nb)
#pragma unroll
    for (int r = 0; r < 8; ++r){
      float v = fmaxf(acc1[nb][r] + acc2[nb][r], 0.f);
      out[(long)(vox0 + half*8 + r)*NCH + nb*16 + n0] = f2bf(v);
    }
}

// ---------------- PSP pooling / head (VALU; negligible FLOPs) ------
__global__ void pool_global(const bf16* __restrict__ act, float* __restrict__ out){
  int c = blockIdx.x;
  float s = 0.f;
  for (long v = threadIdx.x; v < NV; v += 256) s += bf2f(act[v*NCH + c]);
  __shared__ float red[256];
  red[threadIdx.x] = s; __syncthreads();
  for (int st = 128; st > 0; st >>= 1){
    if ((int)threadIdx.x < st) red[threadIdx.x] += red[threadIdx.x + st];
    __syncthreads();
  }
  if (threadIdx.x == 0) out[c] = red[0] / (float)NV;
}

__global__ void pool_window(const bf16* __restrict__ act, float* __restrict__ out,
                            int od, int oh, int ow, int kd, int kh, int kw){
  int idx = blockIdx.x*blockDim.x + threadIdx.x;
  int total = od*oh*ow*NCH;
  if (idx >= total) return;
  int c = idx & 63;
  int v = idx >> 6;
  int x = v % ow; int y = (v / ow) % oh; int z = v / (ow*oh);
  float s = 0.f;
  for (int dz = 0; dz < kd; ++dz)
    for (int dy = 0; dy < kh; ++dy)
      for (int dx = 0; dx < kw; ++dx){
        long vv = ((long)(z*kd + dz)*HH + (y*kh + dy))*WW + (x*kw + dx);
        s += bf2f(act[vv*NCH + c]);
      }
  out[(long)v*NCH + c] = s / (float)(kd*kh*kw);
}

__global__ void chan_mix(const float* __restrict__ in, const float* __restrict__ w,
                         float* __restrict__ out, int nvox){
  int i = blockIdx.x*blockDim.x + threadIdx.x;
  if (i >= nvox*NCH) return;
  int v = i >> 6, o = i & 63;
  const float* a  = in + (long)v*NCH;
  const float* wr = w + o*64;
  float s = 0.f;
  for (int c = 0; c < 64; ++c) s += a[c]*wr[c];
  out[i] = s;
}

struct Lin { int a, b; float f; };
__device__ __forceinline__ Lin mklin(int i, int D, int S){
  float src = (i + 0.5f) * ((float)S/(float)D) - 0.5f;   // align_corners=False
  float fl = floorf(src);
  Lin r; r.f = src - fl;
  int i0 = (int)fl, i1 = i0 + 1;
  r.a = i0 < 0 ? 0 : (i0 > S-1 ? S-1 : i0);
  r.b = i1 < 0 ? 0 : (i1 > S-1 ? S-1 : i1);
  return r;
}
__device__ __forceinline__ float tri3(const float* p, int OH, int OW,
                                      Lin z, Lin y, Lin x, int c){
#define AT(zi,yi,xi) p[(((long)(zi)*OH + (yi))*OW + (xi))*NCH + c]
  float c00 = AT(z.a,y.a,x.a)*(1.f-x.f) + AT(z.a,y.a,x.b)*x.f;
  float c01 = AT(z.a,y.b,x.a)*(1.f-x.f) + AT(z.a,y.b,x.b)*x.f;
  float c10 = AT(z.b,y.a,x.a)*(1.f-x.f) + AT(z.b,y.a,x.b)*x.f;
  float c11 = AT(z.b,y.b,x.a)*(1.f-x.f) + AT(z.b,y.b,x.b)*x.f;
#undef AT
  float c0 = c00*(1.f-y.f) + c01*y.f;
  float c1 = c10*(1.f-y.f) + c11*y.f;
  return c0*(1.f-z.f) + c1*z.f;
}

__global__ void psp_combine(const bf16* __restrict__ x,
                            const float* __restrict__ q0, const float* __restrict__ q1,
                            const float* __restrict__ q2, const float* __restrict__ q3,
                            float* __restrict__ out_f32,  // NCDHW slab of d_out
                            bf16* __restrict__ out_bf){
  int v = blockIdx.x*blockDim.x + threadIdx.x;
  if (v >= NV) return;
  int xw = v & (WW-1), yh = (v >> 7) & (HH-1), zd = v >> 13;
  Lin z1 = mklin(zd, DD,  8), y1 = mklin(yh, HH,  8), x1 = mklin(xw, WW,  8);
  Lin z2 = mklin(zd, DD, 16), y2 = mklin(yh, HH, 16), x2 = mklin(xw, WW, 16);
  Lin z3 = mklin(zd, DD, 24), y3 = mklin(yh, HH, 32), x3 = mklin(xw, WW, 25);
  const bf16* xr = x + (long)v*NCH;
  for (int c = 0; c < NCH; ++c){
    float o = q0[c]
            + tri3(q1,  8,  8, z1, y1, x1, c)
            + tri3(q2, 16, 16, z2, y2, x2, c)
            + tri3(q3, 32, 25, z3, y3, x3, c);
    float s = bf2f(xr[c]) + 0.25f*o;
    float r = fmaxf(s*0.5f, 0.f);
    out_f32[(long)c*NV + v] = r;       // final x output (NCDHW f32)
    out_bf[(long)v*NCH + c] = f2bf(r); // feeds classify head
  }
}

// head tail: 64 -> 1 channel 3x3x3 conv + bias (1.4 GFLOP, VALU fine)
__global__ void conv3_c1(const bf16* __restrict__ act, const float* __restrict__ w,
                         const float* __restrict__ bias, float* __restrict__ out){
  int v = blockIdx.x*blockDim.x + threadIdx.x;
  if (v >= NV) return;
  int x = v & (WW-1), y = (v >> 7) & (HH-1), z = v >> 13;
  float s = bias[0];
  int t = 0;
  for (int dz = -1; dz <= 1; ++dz){
    int zz = z + dz;
    for (int dy = -1; dy <= 1; ++dy){
      int yy = y + dy;
      for (int dx = -1; dx <= 1; ++dx, ++t){
        int xx = x + dx;
        if ((unsigned)zz < (unsigned)DD && (unsigned)yy < (unsigned)HH &&
            (unsigned)xx < (unsigned)WW){
          const bf16* a = act + ((long)(zz*HH + yy)*WW + xx)*NCH;
          float acc = 0.f;
          for (int c = 0; c < NCH; ++c) acc += bf2f(a[c]) * w[c*27 + t];
          s += acc;
        }
      }
    }
  }
  out[v] = s;
}

// ============================================================
extern "C" void kernel_launch(void* const* d_in, const int* in_sizes, int n_in,
                              void* d_out, int out_size, void* d_ws, size_t ws_size,
                              hipStream_t stream) {
  const float* fvl   = (const float*)d_in[0];
  const float* w_a1  = (const float*)d_in[1];
  const float* w_a2  = (const float*)d_in[2];
  const float* w_nin = (const float*)d_in[3];
  const float* w1s   = (const float*)d_in[4];
  const float* w2s   = (const float*)d_in[5];
  const float* wp0   = (const float*)d_in[6];
  const float* wp1   = (const float*)d_in[7];
  const float* wp2   = (const float*)d_in[8];
  const float* wp3   = (const float*)d_in[9];
  const float* wc1   = (const float*)d_in[10];
  const float* wc2   = (const float*)d_in[11];
  const float* bc    = (const float*)d_in[12];
  float* out = (float*)d_out;

  char* p = (char*)d_ws;
  auto bump = [&](size_t bytes)->char* {
    char* r = p; p += (bytes + 255) & ~(size_t)255; return r;
  };
  bf16* buf0 = (bf16*)bump((size_t)NV*NCH*2);   // bf16(fvl), later head temp
  bf16* buf1 = (bf16*)bump((size_t)NV*NCH*2);
  bf16* buf2 = (bf16*)bump((size_t)NV*NCH*2);
  bf16* pkA  = (bf16*)bump(110592*2);           // w_a1 pack (54 ksteps)
  bf16* pkB  = (bf16*)bump(110592*2);           // w_a2
  bf16* pkC  = (bf16*)bump(110592*2);           // wc1
  bf16* pkN  = (bf16*)bump(4096*2);             // w_nin (2 ksteps)
  bf16* pk1  = (bf16*)bump(5*4096*2);
  bf16* pk2  = (bf16*)bump(5*4096*2);
  float* p0raw = (float*)bump(64*4);
  float* p1raw = (float*)bump(512*64*4);
  float* p2raw = (float*)bump(4096*64*4);
  float* p3raw = (float*)bump(19200*64*4);
  float* q0 = (float*)bump(64*4);
  float* q1 = (float*)bump(512*64*4);
  float* q2 = (float*)bump(4096*64*4);
  float* q3 = (float*)bump(19200*64*4);

  // 0) input -> channel-last bf16 ; weight packs
  cvt_in<<<(NV*NCH + 255)/256, 256, 0, stream>>>(fvl, buf0);
  pack_w<<<(110592 + 255)/256, 256, 0, stream>>>(w_a1, pkA, 54, 1);
  pack_w<<<(110592 + 255)/256, 256, 0, stream>>>(w_a2, pkB, 54, 1);
  pack_w<<<(110592 + 255)/256, 256, 0, stream>>>(wc1,  pkC, 54, 1);
  pack_w<<<16, 256, 0, stream>>>(w_nin, pkN, 2, 0);
  for (int i = 0; i < 5; ++i){
    pack_w<<<16, 256, 0, stream>>>(w1s + i*4096, pk1 + i*4096, 2, 0);
    pack_w<<<16, 256, 0, stream>>>(w2s + i*4096, pk2 + i*4096, 2, 0);
  }

  // 1) x = conv3(conv3(fvl,wa1),wa2) + conv1(fvl,wnin)
  conv3_wmma<0><<<NBLK3, 256, 0, stream>>>(buf0, pkA, nullptr, nullptr, buf1);
  conv3_wmma<1><<<NBLK3, 256, 0, stream>>>(buf1, pkB, buf0, pkN, buf2);

  // 2) five fused 1x1 residual blocks (ping-pong buf2/buf1)
  bf16* src = buf2; bf16* dst = buf1;
  for (int i = 0; i < 5; ++i){
    conv1_res_wmma<<<NBLK1, 256, 0, stream>>>(src, pk1 + i*4096, pk2 + i*4096, dst);
    bf16* t = src; src = dst; dst = t;
  }
  // x now in src (== buf1)

  // 3) PSP pyramid: ps = {1,8,16,24} -> pooled (1, 8^3, 16^3, 24x32x25)
  pool_global<<<64, 256, 0, stream>>>(src, p0raw);
  pool_window<<<(512*64 + 255)/256, 256, 0, stream>>>(src, p1raw, 8, 8, 8, 6, 8, 16);
  pool_window<<<(4096*64 + 255)/256, 256, 0, stream>>>(src, p2raw, 16, 16, 16, 3, 4, 8);
  pool_window<<<(19200*64 + 255)/256, 256, 0, stream>>>(src, p3raw, 24, 32, 25, 2, 2, 5);
  chan_mix<<<1, 64, 0, stream>>>(p0raw, wp0, q0, 1);
  chan_mix<<<(512*64 + 255)/256, 256, 0, stream>>>(p1raw, wp1, q1, 512);
  chan_mix<<<(4096*64 + 255)/256, 256, 0, stream>>>(p2raw, wp2, q2, 4096);
  chan_mix<<<(19200*64 + 255)/256, 256, 0, stream>>>(p3raw, wp3, q3, 19200);

  // 4) combine + relu(out_sum/2): f32 x -> d_out[0..64*NV), bf16 x -> buf2
  psp_combine<<<(NV + 255)/256, 256, 0, stream>>>(src, q0, q1, q2, q3, out, buf2);

  // 5) classify head: relu(conv3(x,wc1)) -> buf0 ; conv3(.,wc2)+bc -> d_out tail
  conv3_wmma<2><<<NBLK3, 256, 0, stream>>>(buf2, pkC, nullptr, nullptr, buf0);
  conv3_c1<<<(NV + 255)/256, 256, 0, stream>>>(buf0, wc2, bc, out + (size_t)NCH*NV);
}
